// CausalSelfAttention_70446053589327
// MI455X (gfx1250) — compile-verified, matrix operand order checked by reading, not run on hardware
//
#include <hip/hip_runtime.h>

typedef __attribute__((ext_vector_type(16))) _Float16 v16h;
typedef __attribute__((ext_vector_type(8)))  _Float16 v8h;
typedef __attribute__((ext_vector_type(8)))  float    v8f;
typedef __attribute__((ext_vector_type(4)))  float    v4f;
typedef __attribute__((ext_vector_type(4)))  int      v4i;

union ABFrag { v16h v; v8h h[2]; };

// ---- gfx1250 async global->LDS (ASYNCcnt) with safe fallback --------------
#if defined(__gfx1250__) && __has_builtin(__builtin_amdgcn_global_load_async_to_lds_b128) && __has_builtin(__builtin_amdgcn_s_wait_asynccnt)
#define USE_ASYNC_LDS 1
#else
#define USE_ASYNC_LDS 0
#endif

static __device__ inline void g2lds_16B(const _Float16* g, _Float16* l) {
#if USE_ASYNC_LDS
  __builtin_amdgcn_global_load_async_to_lds_b128(
      (__attribute__((address_space(1))) v4i*)(void*)g,
      (__attribute__((address_space(3))) v4i*)(void*)l, 0, 0);
#else
  *(v8h*)l = *(const v8h*)g;
#endif
}
static __device__ inline void g2lds_wait() {
#if USE_ASYNC_LDS
  __builtin_amdgcn_s_wait_asynccnt(0);
#endif
}

// Load a 16x32 f16 WMMA operand fragment from a row-major [row][k] buffer.
// Works for A (row = M) and for B staged as [N][K] (row = N).
// wave32 layout: lane L -> row = L&15; halves h -> k = (h&7) + 16*(h>>3) + 8*(L>>4)
//  => two contiguous 16B loads at k offsets {8g, 16+8g}.
static __device__ inline v16h load_frag(const _Float16* base, int ld) {
  const int lane = threadIdx.x & 31;
  const int m = lane & 15;
  const int g = lane >> 4;
  const _Float16* p = base + (size_t)m * ld + 8 * g;
  ABFrag f;
  f.h[0] = *(const v8h*)(p);
  f.h[1] = *(const v8h*)(p + 16);
  return f.v;
}

static __device__ inline v8f wmma_f16(v16h a, v16h b, v8f c) {
  return __builtin_amdgcn_wmma_f32_16x16x32_f16(
      /*neg_a=*/false, a, /*neg_b=*/false, b,
      /*c_mod=*/(short)0, c, /*reuse_a=*/false, /*reuse_b=*/false);
}

// Vectorized f32 -> f16 conversion, 8 elements (one b128 store) per step.
__global__ void cvt_f32_f16(const float* __restrict__ in,
                            _Float16* __restrict__ out, int n8) {
  int i = blockIdx.x * blockDim.x + threadIdx.x;
  int stride = gridDim.x * blockDim.x;
  for (; i < n8; i += stride) {
    const v4f* p = (const v4f*)(in + (size_t)i * 8);
    v4f a = p[0], b = p[1];
    v8h o;
#pragma unroll
    for (int j = 0; j < 4; j++) { o[j] = (_Float16)a[j]; o[4 + j] = (_Float16)b[j]; }
    *(v8h*)(out + (size_t)i * 8) = o;
  }
}

// C = A[M,K] * B[K,N] + bias[N].
// Block tile 128x128, 8 waves (2 M x 4 N), wave tile 64x32 = 4x2 WMMA tiles.
// K stepped in 32-chunks through LDS. A staged via async global->LDS b128;
// B staged transposed ([n][k]) so fragment loads are contiguous 16B ds reads.
template <bool OUT_F16>
__global__ __launch_bounds__(256) void gemm_wmma(
    const _Float16* __restrict__ A, const _Float16* __restrict__ Bw,
    const float* __restrict__ bias,
    _Float16* __restrict__ outh, float* __restrict__ outf,
    int M, int N, int K) {
  constexpr int LDSS = 40;  // halves; 80B row stride, 16B aligned
  __shared__ __align__(16) _Float16 ldsA[128 * LDSS];
  __shared__ __align__(16) _Float16 ldsB[128 * LDSS];

  const int tid = threadIdx.x;
  const int lane = tid & 31;
  const int wave = tid >> 5;
  const int wm = wave & 1;   // 0..1 -> 64 rows each
  const int wn = wave >> 1;  // 0..3 -> 32 cols each
  const int n0 = blockIdx.x * 128;
  const int m0 = blockIdx.y * 128;

  v8f acc[4][2];
#pragma unroll
  for (int i = 0; i < 4; i++)
#pragma unroll
    for (int j = 0; j < 2; j++) acc[i][j] = {};

  for (int k0 = 0; k0 < K; k0 += 32) {
    // Stage A tile 128x32: 512 16B chunks, 2 per thread (async -> LDS).
#pragma unroll
    for (int i = 0; i < 2; i++) {
      int c = tid + i * 256;
      int row = c >> 2, cc = (c & 3) * 8;
      g2lds_16B(A + (size_t)(m0 + row) * K + k0 + cc, &ldsA[row * LDSS + cc]);
    }
    // Stage B tile 32x128 transposed into ldsB[n][k]: 512 chunks, 2/thread.
#pragma unroll
    for (int i = 0; i < 2; i++) {
      int c = tid + i * 256;
      int k = c >> 4, cc = (c & 15) * 8;
      v8h t = *(const v8h*)(Bw + (size_t)(k0 + k) * N + n0 + cc);
#pragma unroll
      for (int j = 0; j < 8; j++) ldsB[(cc + j) * LDSS + k] = t[j];
    }
    g2lds_wait();
    __syncthreads();

    v16h af[4], bf[2];
#pragma unroll
    for (int mi = 0; mi < 4; mi++)
      af[mi] = load_frag(&ldsA[(wm * 64 + mi * 16) * LDSS], LDSS);
#pragma unroll
    for (int ni = 0; ni < 2; ni++)
      bf[ni] = load_frag(&ldsB[(wn * 32 + ni * 16) * LDSS], LDSS);
#pragma unroll
    for (int mi = 0; mi < 4; mi++)
#pragma unroll
      for (int ni = 0; ni < 2; ni++)
        acc[mi][ni] = wmma_f16(af[mi], bf[ni], acc[mi][ni]);
    __syncthreads();
  }

  // Epilogue: C/D layout lane -> N = lane&15, vgpr r -> M = r + 8*(lane>>4).
  const int nlo = lane & 15, g = lane >> 4;
#pragma unroll
  for (int mi = 0; mi < 4; mi++)
#pragma unroll
    for (int ni = 0; ni < 2; ni++) {
      int ncol = n0 + wn * 32 + ni * 16 + nlo;
      float bv = bias[ncol];
#pragma unroll
      for (int r = 0; r < 8; r++) {
        int mrow = m0 + wm * 64 + mi * 16 + r + 8 * g;
        float v = acc[mi][ni][r] + bv;
        if (OUT_F16)
          outh[(size_t)mrow * N + ncol] = (_Float16)v;
        else
          outf[(size_t)mrow * N + ncol] = v;
      }
    }
}

// Flash-attention, causal. Grid: B*H*(T/64) blocks of 128 threads (4 waves).
// Each wave owns 16 query rows; s processed in 32-wide blocks.
__global__ __launch_bounds__(128) void attn_wmma(
    const _Float16* __restrict__ qkv, _Float16* __restrict__ y) {
  constexpr int T = 2048, C = 1024, D = 64, LD = 3 * C;
  constexpr int LSV = 40;
  __shared__ __align__(16) _Float16 ldsV[D * LSV];      // [d][s] transposed V block
  __shared__ __align__(16) _Float16 ldsP[4 * 16 * LSV]; // per-wave P tile [m][s]

  const int tid = threadIdx.x;
  const int lane = tid & 31;
  const int wave = tid >> 5;
  const int nlo = lane & 15, g = lane >> 4;

  const int idx = blockIdx.x;
  const int qblk = idx & 31;  // T/64 = 32
  const int bh = idx >> 5;
  const int h = bh & 15;
  const int b = bh >> 4;

  const int q0 = qblk * 64;
  const int qbase = q0 + wave * 16;

  const _Float16* Qp = qkv + (size_t)b * T * LD + (size_t)h * D;
  const _Float16* Kp = Qp + C;
  const _Float16* Vp = Qp + 2 * C;

  v16h qf[2];
#pragma unroll
  for (int dk = 0; dk < 2; dk++)
    qf[dk] = load_frag(Qp + (size_t)qbase * LD + dk * 32, LD);

  v8f oacc[4];
#pragma unroll
  for (int dt = 0; dt < 4; dt++) oacc[dt] = {};
  float mrow[8], lrow[8];
#pragma unroll
  for (int r = 0; r < 8; r++) { mrow[r] = -1e30f; lrow[r] = 0.0f; }

  const float scale = 0.125f;     // 1/sqrt(64)
  const int nblk = q0 / 32 + 2;   // cover s in [0, q0+63]

  _Float16* ldsPw = &ldsP[wave * 16 * LSV];

  for (int j = 0; j < nblk; j++) {
    const int s0 = j * 32;
    __syncthreads();
    // Stage V[s0..s0+32)[0..64) transposed into ldsV[d][s].
#pragma unroll
    for (int i = 0; i < 2; i++) {
      int task = tid + i * 128;
      int s = task >> 3, cc = (task & 7) * 8;
      v8h t = *(const v8h*)(Vp + (size_t)(s0 + s) * LD + cc);
#pragma unroll
      for (int jj = 0; jj < 8; jj++) ldsV[(cc + jj) * LSV + s] = t[jj];
    }
    __syncthreads();

    // S = Q * K^T : K rows are native [n=s][k=d] fragments in global memory.
    v8f sacc[2];
    sacc[0] = {}; sacc[1] = {};
#pragma unroll
    for (int st = 0; st < 2; st++)
#pragma unroll
      for (int dk = 0; dk < 2; dk++) {
        v16h kf = load_frag(Kp + (size_t)(s0 + st * 16) * LD + dk * 32, LD);
        sacc[st] = wmma_f16(qf[dk], kf, sacc[st]);
      }

    // Online softmax; lane owns rows m = r + 8g, row elements across 16 lanes.
#pragma unroll
    for (int r = 0; r < 8; r++) {
      const int qg = qbase + r + 8 * g;
      float v0 = sacc[0][r] * scale;
      float v1 = sacc[1][r] * scale;
      if (s0 + nlo > qg) v0 = -1e30f;
      if (s0 + 16 + nlo > qg) v1 = -1e30f;
      float t = fmaxf(v0, v1);
#pragma unroll
      for (int d = 1; d < 16; d <<= 1) t = fmaxf(t, __shfl_xor(t, d, 16));
      float mnew = fmaxf(mrow[r], t);
      float alpha = __expf(mrow[r] - mnew);
      mrow[r] = mnew;
      float p0 = __expf(v0 - mnew);
      float p1 = __expf(v1 - mnew);
      float rs = p0 + p1;
#pragma unroll
      for (int d = 1; d < 16; d <<= 1) rs += __shfl_xor(rs, d, 16);
      lrow[r] = lrow[r] * alpha + rs;
#pragma unroll
      for (int dt = 0; dt < 4; dt++) oacc[dt][r] *= alpha;
      const int m = r + 8 * g;
      ldsPw[m * LSV + nlo] = (_Float16)p0;
      ldsPw[m * LSV + 16 + nlo] = (_Float16)p1;
    }
    __syncthreads();

    // O += P * V
    v16h pf = load_frag(ldsPw, LSV);
#pragma unroll
    for (int dt = 0; dt < 4; dt++) {
      v16h vf = load_frag(&ldsV[dt * 16 * LSV], LSV);
      oacc[dt] = wmma_f16(pf, vf, oacc[dt]);
    }
  }

  // Normalize and store y[b, q, h*64 + d] as f16.
#pragma unroll
  for (int dt = 0; dt < 4; dt++)
#pragma unroll
    for (int r = 0; r < 8; r++) {
      int m = r + 8 * g;
      int qg = qbase + m;
      float v = oacc[dt][r] / lrow[r];
      y[((size_t)b * T + qg) * C + h * D + dt * 16 + nlo] = (_Float16)v;
    }
}

extern "C" void kernel_launch(void* const* d_in, const int* in_sizes, int n_in,
                              void* d_out, int out_size, void* d_ws, size_t ws_size,
                              hipStream_t stream) {
  (void)in_sizes; (void)n_in; (void)out_size; (void)ws_size;
  const float* x  = (const float*)d_in[0];
  // d_in[1] = attn_mask: causal mask applied analytically in-kernel.
  const float* Wa = (const float*)d_in[2];
  const float* ba = (const float*)d_in[3];
  const float* Wp = (const float*)d_in[4];
  const float* bp = (const float*)d_in[5];
  float* out = (float*)d_out;

  constexpr int B = 2, T = 2048, C = 1024;
  constexpr size_t NX = (size_t)B * T * C;
  constexpr size_t NWA = (size_t)C * 3 * C;
  constexpr size_t NWP = (size_t)C * C;
  constexpr size_t NQKV = (size_t)B * T * 3 * C;

  _Float16* xh   = (_Float16*)d_ws;
  _Float16* Wah  = xh + NX;
  _Float16* Wph  = Wah + NWA;
  _Float16* qkvh = Wph + NWP;
  _Float16* yh   = qkvh + NQKV;  // total ws use: 48 MB

  cvt_f32_f16<<<1024, 256, 0, stream>>>(x, xh, (int)(NX / 8));
  cvt_f32_f16<<<1024, 256, 0, stream>>>(Wa, Wah, (int)(NWA / 8));
  cvt_f32_f16<<<1024, 256, 0, stream>>>(Wp, Wph, (int)(NWP / 8));

  // qkv = x @ W_attn + b_attn  (f16 out)
  dim3 g1(3 * C / 128, (B * T) / 128);
  gemm_wmma<true><<<g1, 256, 0, stream>>>(xh, Wah, ba, qkvh, nullptr,
                                          B * T, 3 * C, C);

  // causal flash attention -> yh (f16)
  attn_wmma<<<B * 16 * (T / 64), 128, 0, stream>>>(qkvh, yh);

  // out = y @ W_proj + b_proj  (f32 out)
  dim3 g2(C / 128, (B * T) / 128);
  gemm_wmma<false><<<g2, 256, 0, stream>>>(yh, Wph, bp, nullptr, out,
                                           B * T, C, C);
}
